// ResNet_16518444220817
// MI455X (gfx1250) — matrix operand order repaired and checked by reading, not compiled, not run on hardware
//
#include <hip/hip_runtime.h>

// ---------------------------------------------------------------------------
// Sparse ResNet block (gather-GEMM-scatter) for MI455X / gfx1250.
// bf16 WMMA (v_wmma_f32_16x16x32_bf16), fp32 accumulate, fp32 epilogue.
// Weights staged per-workgroup into LDS with async global->LDS copies
// (ASYNCcnt), double-buffered across the 27 kernel offsets.
// ---------------------------------------------------------------------------

typedef __attribute__((ext_vector_type(16))) __bf16 v16bf;
typedef __attribute__((ext_vector_type(8)))  __bf16 v8bf;
typedef __attribute__((ext_vector_type(4)))  __bf16 v4bf;
typedef __attribute__((ext_vector_type(8)))  float  v8f;
typedef __attribute__((ext_vector_type(4)))  int    v4i;

#define SP_N   300000
#define SP_C   64
#define SP_K   27
#define TILES  (SP_N / 16)      // 18750 row tiles of 16
#define TPW    3                // row tiles per wave
#define NWAVES (TILES / TPW)    // 6250
#define WPB    25               // waves per block (6250 / 25 = 250 blocks exact)
#define BLK    (WPB * 32)       // 800 threads
#define NBLK   (NWAVES / WPB)   // 250

// ---- prep: x (f32) -> xb (bf16), 4 elements per thread -------------------
__global__ __launch_bounds__(256)
void prep_x_kernel(const float* __restrict__ x, __bf16* __restrict__ xb, int nelem)
{
    int j = (blockIdx.x * blockDim.x + threadIdx.x) * 4;
    if (j + 3 < nelem) {
        float4 v = *(const float4*)(x + j);
        v4bf o;
        o[0] = (__bf16)v.x; o[1] = (__bf16)v.y;
        o[2] = (__bf16)v.z; o[3] = (__bf16)v.w;
        *(v4bf*)(xb + j) = o;
    }
}

// ---- prep: W [K,C,C] f32 -> bf16 B-fragment chunks -----------------------
// Per k: 8 KB block of 512 16-byte chunks. Chunk index within the k-block:
//   c = ((kk*4 + n)*2 + h2)*32 + lane      (h2 = element half of v16bf)
// Element e8 (0..7) of chunk c is B element:
//   c_in  = 32*kk + 16*h2 + 8*(lane>>4) + e8
//   c_out = 16*n + (lane&15)
// so a wave's ds_load_b128 at (chunk_base + lane*16) is stride-16B-by-lane
// (bank-conflict free) and staging W[k] is a flat 8 KB copy.
__global__ __launch_bounds__(256)
void prep_w_kernel(const float* __restrict__ W, __bf16* __restrict__ wb)
{
    int r = blockIdx.x * blockDim.x + threadIdx.x;
    if (r >= SP_K * 4096) return;             // 27*64*64 = 110592 elements
    int e8   = r & 7;
    int lane = (r >> 3) & 31;
    int h2   = (r >> 8) & 1;
    int fn   = (r >> 9) & 7;                  // kk*4 + n
    int k    = r >> 12;
    int n    = fn & 3;
    int kk   = fn >> 2;
    int e    = h2 * 8 + e8;
    int c_in  = kk * 32 + h2 * 16 + (lane >> 4) * 8 + e8;
    int c_out = n * 16 + (lane & 15);
    wb[r] = (__bf16)W[k * (SP_C * SP_C) + c_in * SP_C + c_out];
}

// ---- async global -> LDS staging of one k-offset's weights (8 KB) --------
__device__ __forceinline__ void stage_wk(const __bf16* __restrict__ wfrag,
                                         __bf16* buf, int k, int tid)
{
    if (tid < 512) {                           // 512 chunks x 16 B = 8 KB
        const char* src = (const char*)wfrag + (size_t)k * 8192 + (size_t)tid * 16;
        char*       dst = (char*)buf + (size_t)tid * 16;
#if defined(__gfx1250__) && __has_builtin(__builtin_amdgcn_global_load_async_to_lds_b128)
        __builtin_amdgcn_global_load_async_to_lds_b128(
            (__attribute__((address_space(1))) v4i*)src,
            (__attribute__((address_space(3))) v4i*)dst,
            /*offset=*/0, /*cpol=*/0);
#else
        *(v8bf*)dst = *(const v8bf*)src;       // sync fallback: load + ds_store
#endif
    }
}

__device__ __forceinline__ void wait_async0()
{
#if __has_builtin(__builtin_amdgcn_s_wait_asynccnt)
    __builtin_amdgcn_s_wait_asynccnt(0);
#else
    asm volatile("s_wait_asynccnt 0" ::: "memory");
#endif
}

// ---- main layer: gather + 16x16x32 bf16 WMMA + epilogue ------------------
// MODE 0: out16 = bf16(relu(acc + bias))        (feeds next layer's gather)
// MODE 1: out32 = acc + bias + resid            (final fp32 output)
template<int MODE>
__global__ __launch_bounds__(BLK)
void sconv_layer(const __bf16* __restrict__ feats,
                 const __bf16* __restrict__ wfrag,
                 const float*  __restrict__ bias,
                 const int*    __restrict__ nbr,
                 const float*  __restrict__ resid,
                 float*        __restrict__ out32,
                 __bf16*       __restrict__ out16)
{
    __shared__ __align__(16) __bf16 lds_w[2][4096];   // 2 x 8 KB double buffer

    const int tid  = threadIdx.x;
    const int wid  = blockIdx.x * WPB + (tid >> 5);   // exact: no guard needed
    const int lane = tid & 31;
    const int m = lane & 15;                  // A row within tile / B column
    const int g = lane >> 4;                  // lane half (K sub-block)

    const long row0 = (long)wid * (16 * TPW);

    v8f acc[TPW][4] = {};

    stage_wk(wfrag, &lds_w[0][0], 0, tid);
    wait_async0();
    __syncthreads();

    for (int k = 0; k < SP_K; ++k) {
        if (k + 1 < SP_K)                      // prefetch next k while computing
            stage_wk(wfrag, &lds_w[(k + 1) & 1][0], k + 1, tid);

        // --- gather A fragments for TPW tiles (16-bit A layout) ---
        v16bf a0[TPW], a1[TPW];
        #pragma unroll
        for (int t = 0; t < TPW; ++t) {
            int ridx = nbr[(long)k * SP_N + row0 + t * 16 + m];
            const __bf16* ar = feats + (long)ridx * SP_C;
            v8bf p0 = *(const v8bf*)(ar + 8 * g);        // K 0..7   / 8..15
            v8bf p1 = *(const v8bf*)(ar + 16 + 8 * g);   // K 16..23 / 24..31
            v8bf p2 = *(const v8bf*)(ar + 32 + 8 * g);   // K 32..39 / 40..47
            v8bf p3 = *(const v8bf*)(ar + 48 + 8 * g);   // K 48..55 / 56..63
            a0[t] = __builtin_shufflevector(p0, p1, 0,1,2,3,4,5,6,7,8,9,10,11,12,13,14,15);
            a1[t] = __builtin_shufflevector(p2, p3, 0,1,2,3,4,5,6,7,8,9,10,11,12,13,14,15);
        }

        // --- B fragments from LDS (bank-conflict-free ds_load_b128) ---
        const __bf16* bufk = &lds_w[k & 1][0];
        #pragma unroll
        for (int n = 0; n < 4; ++n) {
            v8bf q0 = *(const v8bf*)(bufk + ((0 * 4 + n) * 2 + 0) * 256 + lane * 8);
            v8bf q1 = *(const v8bf*)(bufk + ((0 * 4 + n) * 2 + 1) * 256 + lane * 8);
            v8bf q2 = *(const v8bf*)(bufk + ((1 * 4 + n) * 2 + 0) * 256 + lane * 8);
            v8bf q3 = *(const v8bf*)(bufk + ((1 * 4 + n) * 2 + 1) * 256 + lane * 8);
            v16bf b0 = __builtin_shufflevector(q0, q1, 0,1,2,3,4,5,6,7,8,9,10,11,12,13,14,15);
            v16bf b1 = __builtin_shufflevector(q2, q3, 0,1,2,3,4,5,6,7,8,9,10,11,12,13,14,15);
            #pragma unroll
            for (int t = 0; t < TPW; ++t) {
                acc[t][n] = __builtin_amdgcn_wmma_f32_16x16x32_bf16(
                    false, a0[t], false, b0, (short)0, acc[t][n], false, false);
                acc[t][n] = __builtin_amdgcn_wmma_f32_16x16x32_bf16(
                    false, a1[t], false, b1, (short)0, acc[t][n], false, false);
            }
        }

        wait_async0();                         // next-k staging landed in LDS
        __syncthreads();                       // all waves done reading buf[k&1]
    }

    // --- epilogue: C/D layout -> row = v + 8*g, col = 16*n + m ---
    #pragma unroll
    for (int t = 0; t < TPW; ++t) {
        const long r0 = row0 + t * 16;
        #pragma unroll
        for (int n = 0; n < 4; ++n) {
            const int col = n * 16 + m;
            const float bn = bias[col];
            #pragma unroll
            for (int v = 0; v < 8; ++v) {
                const long row = r0 + v + 8 * g;
                float val = acc[t][n][v] + bn;
                if (MODE == 0) {
                    out16[row * SP_C + col] = (__bf16)fmaxf(val, 0.0f);
                } else {
                    out32[row * SP_C + col] = val + resid[row * SP_C + col];
                }
            }
        }
    }
}

// ---------------------------------------------------------------------------
extern "C" void kernel_launch(void* const* d_in, const int* in_sizes, int n_in,
                              void* d_out, int out_size, void* d_ws, size_t ws_size,
                              hipStream_t stream) {
    const float* x   = (const float*)d_in[0];   // [N, C]
    const float* W0  = (const float*)d_in[1];   // [K, C, C]
    const float* b0  = (const float*)d_in[2];   // [C]
    const float* W1  = (const float*)d_in[3];   // [K, C, C]
    const float* b1  = (const float*)d_in[4];   // [C]
    const int*   nbr = (const int*)d_in[5];     // [K, N]
    float* out = (float*)d_out;                 // [N, C] fp32

    // workspace layout (all 256B-aligned offsets)
    char* ws = (char*)d_ws;
    __bf16* xb  = (__bf16*)(ws);                 // 38,400,000 B : x in bf16
    __bf16* h   = (__bf16*)(ws + 38400000);      // 38,400,000 B : relu(conv0) bf16
    __bf16* wb0 = (__bf16*)(ws + 76800000);      //    221,184 B : W0 fragments
    __bf16* wb1 = (__bf16*)(ws + 77021184);      //    221,184 B : W1 fragments

    prep_x_kernel<<<18750, 256, 0, stream>>>(x, xb, SP_N * SP_C);
    prep_w_kernel<<<432, 256, 0, stream>>>(W0, wb0);
    prep_w_kernel<<<432, 256, 0, stream>>>(W1, wb1);

    sconv_layer<0><<<NBLK, BLK, 0, stream>>>(xb, wb0, b0, nbr, nullptr, nullptr, h);
    sconv_layer<1><<<NBLK, BLK, 0, stream>>>(h,  wb1, b1, nbr, x, out, nullptr);

    (void)in_sizes; (void)n_in; (void)out_size; (void)ws_size;
}
